// SelfAttention_11544872091912
// MI455X (gfx1250) — matrix operand order repaired and checked by reading, not compiled, not run on hardware
//
#include <hip/hip_runtime.h>
#include <hip/hip_bf16.h>

#define EMBED 128
#define HEADS 4
#define HDIM  32
#define NB    4
#define LSEQ  4096
#define PITCH 4100   // 16-row score strip pitch (+4 pad -> conflict-free column reads)

typedef __attribute__((ext_vector_type(16))) __bf16 v16bf;
typedef __attribute__((ext_vector_type(8)))  __bf16 v8bf;
typedef __attribute__((ext_vector_type(8)))  float  v8f;
typedef __attribute__((ext_vector_type(4)))  float  f4;

// ---------------- Kernel 1: per-head Q/K/V projections (fp32 -> bf16) ----------------
// One wave (32 lanes) per (n,l,h) row; lane e computes y[e] = sum_d x[d]*W[e][d].
// V is stored transposed [N,H,D,L] so P@V B-fragments are contiguous 32B loads.
__global__ __launch_bounds__(256) void proj_kernel(
    const float* __restrict__ vals, const float* __restrict__ keys,
    const float* __restrict__ qry,
    const float* __restrict__ Wv, const float* __restrict__ Wk,
    const float* __restrict__ Wq,
    __bf16* __restrict__ Qb, __bf16* __restrict__ Kb, __bf16* __restrict__ VbT) {
  __shared__ float wq[HDIM*HDIM], wk[HDIM*HDIM], wv[HDIM*HDIM];
  for (int i = threadIdx.x; i < HDIM*HDIM; i += blockDim.x) {
    wq[i] = Wq[i]; wk[i] = Wk[i]; wv[i] = Wv[i];
  }
  __syncthreads();
  const int lane = threadIdx.x & 31;
  const int row  = blockIdx.x * 8 + (threadIdx.x >> 5);   // over N*L*H
  const int n = row / (LSEQ*HEADS);
  const int rem = row % (LSEQ*HEADS);
  const int l = rem / HEADS;
  const int h = rem % HEADS;
  const long base = ((long)n*LSEQ + l)*EMBED + h*HDIM;
  float xq = qry[base + lane], xk = keys[base + lane], xv = vals[base + lane];
  float yq = 0.f, yk = 0.f, yv = 0.f;
#pragma unroll
  for (int d = 0; d < HDIM; ++d) {
    yq += __shfl(xq, d, 32) * wq[lane*HDIM + d];
    yk += __shfl(xk, d, 32) * wk[lane*HDIM + d];
    yv += __shfl(xv, d, 32) * wv[lane*HDIM + d];
  }
  const long nh = (long)n*HEADS + h;
  Qb [(nh*LSEQ + l)*HDIM + lane] = (__bf16)yq;
  Kb [(nh*LSEQ + l)*HDIM + lane] = (__bf16)yk;
  VbT[(nh*HDIM + lane)*LSEQ + l] = (__bf16)yv;
}

// ---------------- Kernel 2: attention for one (n, h, 16-row q strip) ----------------
// 8 waves / 256 threads; ~280 KB LDS (1 block per WGP, 2 waves per SIMD32).
__global__ __launch_bounds__(256) void attn_kernel(
    const __bf16* __restrict__ Qb, const __bf16* __restrict__ Kb,
    const __bf16* __restrict__ VbT, const int* __restrict__ mask,
    float* __restrict__ attn_out /* [N,H,L,L] */,
    float* __restrict__ Ofull   /* [N,L,EMBED] */) {
  extern __shared__ float smem[];
  float* sS     = smem;                 // 16*PITCH score strip (262.4 KB)
  float* red    = sS + 16*PITCH;        // 256 partials (16 rows x 16)
  float* rowmax = red + 256;            // 16
  float* rowinv = rowmax + 16;          // 16
  float* opart  = rowinv + 16;          // 8*16*32 wave partials of O (16 KB)

  const int qt = blockIdx.x, h = blockIdx.y, n = blockIdx.z;
  const long nh = (long)n*HEADS + h;
  const int tid  = threadIdx.x;
  const int lane = tid & 31;
  const int wave = tid >> 5;            // 0..7
  const int q0 = qt * 16;
  const int hi = lane >> 4;             // lane[4]
  const int lo = lane & 15;
  const int koff = hi * 8;              // unused; A fragment uses contiguous k = hi*16 + e

  // A fragment: Q tile, 16(M) x 32(K) bf16, fixed for the whole block.
  v16bf aq;
  {
    const __bf16* qp = Qb + (nh*LSEQ + (q0 + lo)) * HDIM;
#pragma unroll
    for (int e = 0; e < 16; ++e) {
      const int k = hi*16 + e;
      aq[e] = qp[k];
    }
  }

  // ---- Phase 1: S = Q K^T (masked, pre-scale) into LDS strip ----
  for (int kt = wave; kt < LSEQ/16; kt += 8) {
    const int col = kt*16 + lo;                             // key index (B column)
    const __bf16* kp = Kb + (nh*LSEQ + col)*HDIM + hi*16;   // 32B-aligned
    if (kt + 8 < LSEQ/16)
      __builtin_prefetch(Kb + (nh*LSEQ + col + 128)*HDIM, 0, 1);
    const v8bf t0 = ((const v8bf*)kp)[0];
    const v8bf t1 = ((const v8bf*)kp)[1];
    v16bf bk;
#pragma unroll
    for (int e = 0; e < 8; ++e) { bk[e] = t0[e]; bk[8+e] = t1[e]; }
    v8f c = {};
    c = __builtin_amdgcn_wmma_f32_16x16x32_bf16(false, aq, false, bk,
                                                (short)0, c, false, false);
    const int mv = mask[(long)n*LSEQ + col];
#pragma unroll
    for (int r = 0; r < 8; ++r)
      sS[(r + hi*8)*PITCH + col] = mv ? c[r] : -1e20f;      // mask BEFORE scaling
  }
  __syncthreads();

  // ---- Phase 2: softmax over each row (scale = 1/sqrt(EMBED)) ----
  const float scale = 0.08838834764831845f;                 // 1/sqrt(128)
  {
    const int r = tid >> 4, j = tid & 15;
    float m = -3.0e38f;
    for (int c = j; c < LSEQ; c += 16) m = fmaxf(m, sS[r*PITCH + c]);
    red[r*16 + j] = m;
  }
  __syncthreads();
  if (tid < 16) {
    float m = red[tid*16];
#pragma unroll
    for (int j = 1; j < 16; ++j) m = fmaxf(m, red[tid*16 + j]);
    rowmax[tid] = m * scale;
  }
  __syncthreads();
  {
    const int r = tid >> 4, j = tid & 15;
    const float m = rowmax[r];
    float s = 0.f;
    for (int c = j; c < LSEQ; c += 16) {
      const float p = __expf(sS[r*PITCH + c]*scale - m);
      sS[r*PITCH + c] = p;
      s += p;
    }
    red[r*16 + j] = s;
  }
  __syncthreads();
  if (tid < 16) {
    float s = 0.f;
#pragma unroll
    for (int j = 0; j < 16; ++j) s += red[tid*16 + j];
    rowinv[tid] = 1.0f / s;
  }
  __syncthreads();

  // ---- Phase 3: normalize in LDS; stream attn_w out with NT hint ----
  // (1.07 GB total streaming store: non-temporal keeps K/V resident in L2.)
  {
    float* dst = attn_out + (nh*LSEQ + q0) * (long)LSEQ;
    for (int i = tid; i < 16*LSEQ/4; i += 256) {
      const int idx4 = i * 4;
      const int r = idx4 >> 12;
      const int c = idx4 & (LSEQ - 1);
      const float inv = rowinv[r];
      f4 p = *(f4*)&sS[r*PITCH + c];
      p *= inv;
      *(f4*)&sS[r*PITCH + c] = p;
      __builtin_nontemporal_store(p, (f4*)&dst[(long)r*LSEQ + c]);
    }
  }
  __syncthreads();

  // ---- Phase 4: O = P @ V via WMMA over 32-key slices ----
  v8f c0 = {}, c1 = {};
  for (int s = wave; s < LSEQ/32; s += 8) {
    v16bf ap;
#pragma unroll
    for (int e = 0; e < 16; ++e) {
      const int k = s*32 + hi*16 + e;
      ap[e] = (__bf16)sS[lo*PITCH + k];
    }
    const __bf16* vp0 = VbT + (nh*HDIM +  0 + lo)*LSEQ + s*32 + hi*16;
    const __bf16* vp1 = VbT + (nh*HDIM + 16 + lo)*LSEQ + s*32 + hi*16;
    const v8bf a0 = ((const v8bf*)vp0)[0], a1 = ((const v8bf*)vp0)[1];
    const v8bf b0v = ((const v8bf*)vp1)[0], b1v = ((const v8bf*)vp1)[1];
    v16bf b0, b1;
#pragma unroll
    for (int e = 0; e < 8; ++e) {
      b0[e] = a0[e];  b0[8+e] = a1[e];
      b1[e] = b0v[e]; b1[8+e] = b1v[e];
    }
    c0 = __builtin_amdgcn_wmma_f32_16x16x32_bf16(false, ap, false, b0,
                                                 (short)0, c0, false, false);
    c1 = __builtin_amdgcn_wmma_f32_16x16x32_bf16(false, ap, false, b1,
                                                 (short)0, c1, false, false);
  }
#pragma unroll
  for (int r = 0; r < 8; ++r) {
    const int rowm = r + hi*8;
    opart[wave*512 + rowm*32 + lo]      = c0[r];
    opart[wave*512 + rowm*32 + 16 + lo] = c1[r];
  }
  __syncthreads();
  for (int i = tid; i < 16*32; i += 256) {
    float o = 0.f;
#pragma unroll
    for (int w = 0; w < 8; ++w) o += opart[w*512 + i];
    const int r = i >> 5, d = i & 31;
    Ofull[((long)n*LSEQ + q0 + r)*EMBED + h*HDIM + d] = o;
  }
}

// ---------------- Kernel 3: out = Ofull @ Wo^T + bo ----------------
// Wo (64 KB) staged transposed in LDS once per block; 64 rows amortize it.
#define ROWS_PER_BLK 64
__global__ __launch_bounds__(EMBED) void outproj_kernel(
    const float* __restrict__ Ofull, const float* __restrict__ Wo,
    const float* __restrict__ bo, float* __restrict__ out) {
  extern __shared__ float lds[];
  float* woT = lds;               // [128][128] transposed: woT[j*128+e] = Wo[e*128+j]
  float* x   = woT + EMBED*EMBED; // one input row
  const int e = threadIdx.x;
  // coalesced read of Wo row e' per iteration, scatter-transpose into LDS
  for (int r = 0; r < EMBED; ++r) woT[e*EMBED + r] = Wo[r*EMBED + e];
  const float be = bo[e];
  __syncthreads();
  const long row0 = (long)blockIdx.x * ROWS_PER_BLK;
  for (int rr = 0; rr < ROWS_PER_BLK; ++rr) {
    const long row = row0 + rr;
    x[e] = Ofull[row*EMBED + e];
    __syncthreads();
    float y = be;
#pragma unroll 8
    for (int j = 0; j < EMBED; ++j) y += x[j] * woT[j*EMBED + e];
    out[row*EMBED + e] = y;
    __syncthreads();
  }
}

extern "C" void kernel_launch(void* const* d_in, const int* in_sizes, int n_in,
                              void* d_out, int out_size, void* d_ws, size_t ws_size,
                              hipStream_t stream) {
  (void)in_sizes; (void)n_in; (void)out_size; (void)ws_size;
  const float* vals = (const float*)d_in[0];
  const float* keys = (const float*)d_in[1];
  const float* qry  = (const float*)d_in[2];
  const int*   mask = (const int*)  d_in[3];
  const float* Wv   = (const float*)d_in[4];
  const float* Wk   = (const float*)d_in[5];
  const float* Wq   = (const float*)d_in[6];
  const float* Wo   = (const float*)d_in[7];
  const float* bo   = (const float*)d_in[8];

  float* out  = (float*)d_out;                         // [N,L,EMBED]
  float* attn = out + (size_t)NB*LSEQ*EMBED;           // [N,H,L,L]

  char* ws = (char*)d_ws;
  const size_t seg = (size_t)NB*HEADS*LSEQ*HDIM*sizeof(__bf16);   // 4 MB
  __bf16* Qb  = (__bf16*)(ws);
  __bf16* Kb  = (__bf16*)(ws + seg);
  __bf16* VbT = (__bf16*)(ws + 2*seg);
  float*  Ofl = (float*)(ws + 3*seg);                  // 8 MB [N,L,EMBED]

  proj_kernel<<< (NB*LSEQ*HEADS)/8, 256, 0, stream >>>(vals, keys, qry,
                                                       Wv, Wk, Wq, Qb, Kb, VbT);

  dim3 grid2(LSEQ/16, HEADS, NB);
  const size_t shmem = (size_t)(16*PITCH + 256 + 16 + 16 + 8*16*32) * sizeof(float);
  attn_kernel<<< grid2, 256, shmem, stream >>>(Qb, Kb, VbT, mask, attn, Ofl);

  const size_t shmem3 = (size_t)(EMBED*EMBED + EMBED) * sizeof(float);
  outproj_kernel<<< (NB*LSEQ)/ROWS_PER_BLK, EMBED, shmem3, stream >>>(Ofl, Wo, bo, out);
}
